// GAT_13400297964010
// MI455X (gfx1250) — matrix operand order repaired and checked by reading, not compiled, not run on hardware
//
#include <hip/hip_runtime.h>

// ---------------------------------------------------------------------------
// 2-layer GAT for MI455X (gfx1250, wave32).
// GEMMs: LDS-staged tiles (async global->LDS ASYNCcnt path) feeding
// v_wmma_f32_16x16x32_f16. Edge phase: bandwidth/atomic bound scatter ops.
// ---------------------------------------------------------------------------

typedef _Float16 half8  __attribute__((ext_vector_type(8)));
typedef _Float16 half16 __attribute__((ext_vector_type(16)));
typedef float    float8 __attribute__((ext_vector_type(8)));
typedef int      int4v  __attribute__((ext_vector_type(4)));

#define NEG_SLOPE 0.2f

#if __has_builtin(__builtin_amdgcn_global_load_async_to_lds_b128) && \
    __has_builtin(__builtin_amdgcn_s_wait_asynccnt)
#define USE_ASYNC_LDS 1
#else
#define USE_ASYNC_LDS 0
#endif

// 16-byte global -> LDS copy (async ASYNCcnt-path when the builtin exists).
// Builtin signature (probe-confirmed): (int4 AS1*, int4 AS3*, imm off, imm cpol)
__device__ __forceinline__ void cp16(const _Float16* g, _Float16* l) {
#if USE_ASYNC_LDS
    __builtin_amdgcn_global_load_async_to_lds_b128(
        (__attribute__((address_space(1))) int4v*)(void*)g,
        (__attribute__((address_space(3))) int4v*)(void*)l, 0, 0);
#else
    *(half8*)l = *(const half8*)g;
#endif
}

// Order-preserving map float -> uint so atomicMax(u32) == float max.
__device__ __forceinline__ unsigned encf(float f) {
    unsigned u = __float_as_uint(f);
    return (u & 0x80000000u) ? ~u : (u | 0x80000000u);
}
__device__ __forceinline__ float decf(unsigned u) {
    unsigned v = (u & 0x80000000u) ? (u & 0x7fffffffu) : ~u;
    return __uint_as_float(v);
}

// ---------------------------------------------------------------------------
__global__ void gat_cvt_f16(const float* __restrict__ in, _Float16* __restrict__ out, int n) {
    int i = blockIdx.x * blockDim.x + threadIdx.x;
    if (i < n) out[i] = (_Float16)in[i];
}

// ---------------------------------------------------------------------------
// WMMA GEMM: C[M,NCOLS] = A[M,K] x B[K,NCOLS], f16 inputs, f32 accumulate.
// Block = WPB waves -> 16 rows x 16*WPB cols. A/B panels staged in LDS.
// Fragment layouts per CDNA5 ISA 7.12.2 (A rows striped in-lane; B column n in
// lane n%16, k in A's order at stride LDB; C/D VGPR r -> rows r / r+8).
// ---------------------------------------------------------------------------
template <int K, int WPB, int NCOLS>
__global__ __launch_bounds__(32 * WPB)
void gat_wmma_gemm(const _Float16* __restrict__ A,
                   const _Float16* __restrict__ B,
                   float* __restrict__ C, int M) {
    constexpr int BN  = 16 * WPB;   // block col width
    constexpr int LDA = K + 8;      // padded LDS strides (bank spread)
    constexpr int LDB = BN + 8;
    __shared__ __align__(32) _Float16 sA[16 * LDA];
    __shared__ __align__(32) _Float16 sB[K * LDB];

    const int tid     = threadIdx.x;
    const int lane    = tid & 31;
    const int wave    = tid >> 5;
    const int rowTile = blockIdx.x << 4;
    const int colBase = blockIdx.y * BN;
    const bool fullTile = (rowTile + 16 <= M);   // uniform (scalar) predicate

    // ---- stage A panel (16 x K) ----
    constexpr int KW = K / 8;                    // 16-byte granules per row
    for (int g = tid; g < 16 * KW; g += 32 * WPB) {
        int r   = g / KW;
        int kc  = (g - r * KW) * 8;
        int row = rowTile + r; if (row >= M) row = M - 1;   // tail clamp
        cp16(A + (size_t)row * K + kc, sA + r * LDA + kc);
    }
    // ---- stage B panel (K x BN) ----
    constexpr int CW = BN / 8;
    for (int g = tid; g < K * CW; g += 32 * WPB) {
        int kr = g / CW;
        int cc = (g - kr * CW) * 8;
        cp16(B + (size_t)kr * NCOLS + colBase + cc, sB + kr * LDB + cc);
    }
#if USE_ASYNC_LDS
    __builtin_amdgcn_s_wait_asynccnt(0);
#endif
    __syncthreads();

    const int m        = lane & 15;
    const int kgrp     = (lane < 16) ? 0 : 8;
    const int colLocal = wave << 4;

    float8 acc = {0.f, 0.f, 0.f, 0.f, 0.f, 0.f, 0.f, 0.f};
#pragma unroll
    for (int kt = 0; kt < K; kt += 32) {
        const _Float16* ap = sA + m * LDA + kt + kgrp;
        half16 afrag;
        ((half8*)&afrag)[0] = *(const half8*)ap;
        ((half8*)&afrag)[1] = *(const half8*)(ap + 16);

        const _Float16* bp = sB + (kt + kgrp) * LDB + colLocal + (lane & 15);
        half16 bfrag;
#pragma unroll
        for (int s = 0; s < 16; ++s) bfrag[s] = bp[(s < 8 ? s : s + 8) * LDB];

        acc = __builtin_amdgcn_wmma_f32_16x16x32_f16(
            /*neg_a=*/false, afrag, /*neg_b=*/false, bfrag,
            /*c_mod=*/(short)0, acc, /*reuse_a=*/false, /*reuse_b=*/false);
    }

    const int n = lane & 15;
    const int mbase = rowTile + ((lane < 16) ? 0 : 8);
    float* cp = C + (size_t)mbase * NCOLS + colBase + colLocal + n;
    if (fullTile) {                   // scalar branch, unguarded stores
#pragma unroll
        for (int r = 0; r < 8; ++r) cp[(size_t)r * NCOLS] = acc[r];
    } else {
#pragma unroll
        for (int r = 0; r < 8; ++r)
            if (mbase + r < M) cp[(size_t)r * NCOLS] = acc[r];
    }
}

// ---------------------------------------------------------------------------
// alpha_s[n,h] = sum_c h[n,h,c]*a_src[h,c]; alpha_d likewise. Wave per (n,h).
// ---------------------------------------------------------------------------
template <int H, int C>
__global__ void gat_alpha(const float* __restrict__ hmat,
                          const float* __restrict__ a_src,
                          const float* __restrict__ a_dst,
                          float* __restrict__ out_s, float* __restrict__ out_d,
                          int n) {
    int gw   = (blockIdx.x * blockDim.x + threadIdx.x) >> 5;
    int lane = threadIdx.x & 31;
    if (gw >= n * H) return;
    int node = gw / H, head = gw - node * H;
    float vs = 0.f, vd = 0.f;
#pragma unroll
    for (int c = lane; c < C; c += 32) {
        float hv = hmat[(size_t)node * (H * C) + head * C + c];
        vs += hv * a_src[head * C + c];
        vd += hv * a_dst[head * C + c];
    }
#pragma unroll
    for (int off = 16; off > 0; off >>= 1) {
        vs += __shfl_xor(vs, off, 32);
        vd += __shfl_xor(vd, off, 32);
    }
    if (lane == 0) { out_s[gw] = vs; out_d[gw] = vd; }
}

// ---------------------------------------------------------------------------
// Edge logits: e = leaky_relu(as[src]+ad[dst]); store e; atomicMax per (dst,h).
// Edges [0,E0) from edge_index; edges [E0,E0+n) are self-loops.
// ---------------------------------------------------------------------------
template <int H>
__global__ void gat_edge_logits(const int* __restrict__ src, const int* __restrict__ dst,
                                const float* __restrict__ as, const float* __restrict__ ad,
                                float* __restrict__ ew, unsigned* __restrict__ mbuf,
                                int E0, int n) {
    int tid = blockIdx.x * blockDim.x + threadIdx.x;
    if (tid >= (E0 + n) * H) return;
    int e = tid / H, h = tid - e * H;
    int s, d;
    if (e < E0) { s = src[e]; d = dst[e]; } else { s = d = e - E0; }
    float v = as[s * H + h] + ad[d * H + h];
    v = (v > 0.f) ? v : NEG_SLOPE * v;
    ew[tid] = v;
    atomicMax(mbuf + d * H + h, encf(v));
}

// w = exp(e - max[dst]); overwrite ew; accumulate denominator.
template <int H>
__global__ void gat_edge_exp(const int* __restrict__ dst,
                             float* __restrict__ ew, const unsigned* __restrict__ mbuf,
                             float* __restrict__ denom, int E0, int n) {
    int tid = blockIdx.x * blockDim.x + threadIdx.x;
    if (tid >= (E0 + n) * H) return;
    int e = tid / H, h = tid - e * H;
    int d = (e < E0) ? dst[e] : (e - E0);
    float w = __expf(ew[tid] - decf(mbuf[d * H + h]));
    ew[tid] = w;
    atomicAdd(denom + d * H + h, w);
}

// Wave per edge: agg[dst, :] += h[src, :] * (w / denom[dst])  (per head).
template <int H, int C>
__global__ void gat_edge_aggregate(const int* __restrict__ src, const int* __restrict__ dst,
                                   const float* __restrict__ hmat,
                                   const float* __restrict__ ew,
                                   const float* __restrict__ denom,
                                   float* __restrict__ agg,
                                   int E0, int n) {
    int gw   = (blockIdx.x * blockDim.x + threadIdx.x) >> 5;
    int lane = threadIdx.x & 31;
    if (gw >= E0 + n) return;
    int s, d;
    if (gw < E0) { s = src[gw]; d = dst[gw]; } else { s = d = gw - E0; }
    constexpr int F = H * C;
#pragma unroll
    for (int c = lane; c < F; c += 32) {
        int h = c / C;
        float alpha = ew[(size_t)gw * H + h] / denom[d * H + h];
        float v = hmat[(size_t)s * F + c] * alpha;
        atomicAdd(agg + (size_t)d * F + c, v);
    }
}

// out_f16 = relu(agg + bias)   (layer-1 epilogue, feeds layer-2 GEMM)
__global__ void gat_finalize_relu_f16(const float* __restrict__ agg,
                                      const float* __restrict__ bias,
                                      _Float16* __restrict__ out, int n, int F) {
    int i = blockIdx.x * blockDim.x + threadIdx.x;
    if (i >= n * F) return;
    float v = agg[i] + bias[i % F];
    out[i] = (_Float16)(v > 0.f ? v : 0.f);
}

// out += bias   (layer-2 epilogue; mean over 1 head is identity)
__global__ void gat_add_bias(float* __restrict__ out, const float* __restrict__ bias,
                             int n, int F) {
    int i = blockIdx.x * blockDim.x + threadIdx.x;
    if (i >= n * F) return;
    out[i] += bias[i % F];
}

// ---------------------------------------------------------------------------
static inline int cdiv(long long a, long long b) { return (int)((a + b - 1) / b); }

extern "C" void kernel_launch(void* const* d_in, const int* in_sizes, int n_in,
                              void* d_out, int out_size, void* d_ws, size_t ws_size,
                              hipStream_t stream) {
    const int IN_CH = 128, HID = 32, HEADS = 4, OUT_CH = 32;
    const int F1 = HEADS * HID;                 // 128

    const float* x      = (const float*)d_in[0];
    const int*   ei     = (const int*)d_in[1];
    const float* W1     = (const float*)d_in[2];
    const float* a_src1 = (const float*)d_in[3];
    const float* a_dst1 = (const float*)d_in[4];
    const float* b1     = (const float*)d_in[5];
    const float* W2     = (const float*)d_in[6];
    const float* a_src2 = (const float*)d_in[7];
    const float* a_dst2 = (const float*)d_in[8];
    const float* b2     = (const float*)d_in[9];

    const int Nn = in_sizes[0] / IN_CH;         // 100000
    const int E0 = in_sizes[1] / 2;             // 1600000
    const int ET = E0 + Nn;                     // edges incl. self-loops
    const int* src = ei;
    const int* dst = ei + E0;

    // --- workspace bump allocator (256B aligned regions) ---
    char* ws = (char*)d_ws;
    size_t off = 0;
    auto alloc = [&](size_t bytes) -> char* {
        char* p = ws + off;
        off = (off + bytes + 255) & ~((size_t)255);
        return p;
    };
    _Float16* xh   = (_Float16*)alloc((size_t)Nn * IN_CH * 2);
    _Float16* w1h  = (_Float16*)alloc((size_t)IN_CH * F1 * 2);
    _Float16* w2h  = (_Float16*)alloc((size_t)F1 * OUT_CH * 2);
    float*    h1   = (float*)   alloc((size_t)Nn * F1 * 4);
    float*    s1   = (float*)   alloc((size_t)Nn * HEADS * 4);
    float*    d1   = (float*)   alloc((size_t)Nn * HEADS * 4);
    float*    ew   = (float*)   alloc((size_t)ET * HEADS * 4);
    unsigned* m1   = (unsigned*)alloc((size_t)Nn * HEADS * 4);
    float*    den1 = (float*)   alloc((size_t)Nn * HEADS * 4);
    float*    agg1 = (float*)   alloc((size_t)Nn * F1 * 4);
    // layer-2 aliases (lifetimes disjoint with their layer-1 owners)
    _Float16* h1f16 = xh;       // relu(layer1) in f16; xh dead after GEMM1
    float*    h2    = h1;       // [Nn,32]; h1 dead after layer-1 aggregate
    float*    s2    = s1;
    float*    d2    = d1;
    float*    ew2   = ew;
    unsigned* m2    = m1;
    float*    den2  = den1;
    float*    out   = (float*)d_out;            // [Nn,32] accumulator

    const int B = 256;

    // ===================== Layer 1 =====================
    (void)hipMemsetAsync(agg1, 0, (size_t)Nn * F1 * 4, stream);
    (void)hipMemsetAsync(m1,   0, (size_t)Nn * HEADS * 4, stream); // enc-space min
    (void)hipMemsetAsync(den1, 0, (size_t)Nn * HEADS * 4, stream);

    gat_cvt_f16<<<cdiv((long long)Nn * IN_CH, B), B, 0, stream>>>(x,  xh,  Nn * IN_CH);
    gat_cvt_f16<<<cdiv(IN_CH * F1, B),            B, 0, stream>>>(W1, w1h, IN_CH * F1);
    gat_cvt_f16<<<cdiv(F1 * OUT_CH, B),           B, 0, stream>>>(W2, w2h, F1 * OUT_CH);

    // h1 = x @ W1 : block = 4 waves covering 64 cols, grid.y = 2 -> 128 cols
    gat_wmma_gemm<128, 4, 128><<<dim3(cdiv(Nn, 16), 2), 128, 0, stream>>>(xh, w1h, h1, Nn);

    gat_alpha<4, 32><<<cdiv((long long)Nn * HEADS * 32, B), B, 0, stream>>>(
        h1, a_src1, a_dst1, s1, d1, Nn);

    gat_edge_logits<4><<<cdiv((long long)ET * HEADS, B), B, 0, stream>>>(
        src, dst, s1, d1, ew, m1, E0, Nn);
    gat_edge_exp<4><<<cdiv((long long)ET * HEADS, B), B, 0, stream>>>(
        dst, ew, m1, den1, E0, Nn);
    gat_edge_aggregate<4, 32><<<cdiv((long long)ET * 32, B), B, 0, stream>>>(
        src, dst, h1, ew, den1, agg1, E0, Nn);

    gat_finalize_relu_f16<<<cdiv((long long)Nn * F1, B), B, 0, stream>>>(
        agg1, b1, h1f16, Nn, F1);

    // ===================== Layer 2 =====================
    (void)hipMemsetAsync(m2,   0, (size_t)Nn * 4, stream);
    (void)hipMemsetAsync(den2, 0, (size_t)Nn * 4, stream);
    (void)hipMemsetAsync(out,  0, (size_t)Nn * OUT_CH * 4, stream);

    // h2 = relu(h1) @ W2 : block = 2 waves covering 32 cols
    gat_wmma_gemm<128, 2, 32><<<dim3(cdiv(Nn, 16), 1), 64, 0, stream>>>(h1f16, w2h, h2, Nn);

    gat_alpha<1, 32><<<cdiv((long long)Nn * 32, B), B, 0, stream>>>(
        h2, a_src2, a_dst2, s2, d2, Nn);

    gat_edge_logits<1><<<cdiv((long long)ET, B), B, 0, stream>>>(
        src, dst, s2, d2, ew2, m2, E0, Nn);
    gat_edge_exp<1><<<cdiv((long long)ET, B), B, 0, stream>>>(
        dst, ew2, m2, den2, E0, Nn);
    gat_edge_aggregate<1, 32><<<cdiv((long long)ET * 32, B), B, 0, stream>>>(
        src, dst, h2, ew2, den2, out, E0, Nn);

    gat_add_bias<<<cdiv((long long)Nn * OUT_CH, B), B, 0, stream>>>(out, b2, Nn, OUT_CH);
}
